// RNNModel_44014824849513
// MI455X (gfx1250) — compile-verified, matrix operand order checked by reading, not run on hardware
//
#include <hip/hip_runtime.h>
#include <hip/hip_bf16.h>

// ---------------------------------------------------------------------------
// GRU language model on MI455X (gfx1250), bf16 WMMA + f32 accumulate,
// TDM (tensor_load_to_lds) staging for the decode GEMM A-tile.
// V=32000, E=H=512, T=64, B=64.
// ---------------------------------------------------------------------------

typedef __bf16 bf16_t;
typedef __attribute__((ext_vector_type(16))) __bf16 v16bf;
typedef __attribute__((ext_vector_type(8)))  __bf16 v8bf;
typedef __attribute__((ext_vector_type(8)))  float  v8f;
typedef __attribute__((ext_vector_type(4)))  unsigned int u32x4;
typedef __attribute__((ext_vector_type(4)))  int i32x4;
typedef __attribute__((ext_vector_type(8)))  int i32x8;

#define V_ 32000
#define E_ 512
#define H_ 512
#define T_ 64
#define B_ 64

#if defined(__has_builtin)
#if __has_builtin(__builtin_amdgcn_tensor_load_to_lds) && \
    __has_builtin(__builtin_amdgcn_s_wait_tensorcnt)
#define HAS_TDM 1
#endif
#endif

static __device__ __forceinline__ v8f wmma_bf16(v16bf a, v16bf b, v8f c) {
  // D = A(16x32) * B(32x16) + C(16x16 f32)
  return __builtin_amdgcn_wmma_f32_16x16x32_bf16(
      /*neg_a=*/false, a, /*neg_b=*/false, b,
      /*c_mod=*/(short)0, c, /*reuse_a=*/false, /*reuse_b=*/false);
}

// Load one 16x32 bf16 fragment from a row-major matrix with leading dim `ld`
// (elements). Layout per ISA 7.12.2: lane l -> row (l&15), K-base
// (l&16 ? 8 : 0); v16 elems 0..7 = K+0..7, elems 8..15 = K+16..23.
static __device__ __forceinline__ v16bf load_frag(const bf16_t* base, int ld,
                                                  int lane, int k0) {
  const int r  = lane & 15;
  const int kb = k0 + ((lane & 16) ? 8 : 0);
  const bf16_t* p = base + r * ld + kb;
  v8bf lo = *(const v8bf*)(p);
  v8bf hi = *(const v8bf*)(p + 16);
  v16bf f;
#pragma unroll
  for (int i = 0; i < 8; ++i) { f[i] = lo[i]; f[i + 8] = hi[i]; }
  return f;
}

static __device__ __forceinline__ float sigmoidf_(float x) {
  return 1.0f / (1.0f + __expf(-x));
}

// ------------------------------- prep kernels ------------------------------

__global__ void k_f32_to_bf16(const float* __restrict__ s,
                              bf16_t* __restrict__ d, int n) {
  int i = blockIdx.x * blockDim.x + threadIdx.x;
  int stride = gridDim.x * blockDim.x;
  for (; i < n; i += stride) d[i] = (bf16_t)s[i];
}

__global__ void k_gather_seq(const float* __restrict__ emb,
                             const int* __restrict__ seq,
                             bf16_t* __restrict__ dst) {
  int row = blockIdx.x;                       // 0..T*B-1  (t-major like outs)
  long long v = (long long)seq[row];
  const float* src = emb + v * E_;
  for (int e = threadIdx.x; e < E_; e += blockDim.x)
    dst[row * E_ + e] = (bf16_t)src[e];
}

__global__ void k_gather_word(const float* __restrict__ emb,
                              const int* __restrict__ word,
                              bf16_t* __restrict__ wb, float* __restrict__ wf) {
  int b = blockIdx.x;                         // 0..B-1
  long long v = (long long)word[b];
  const float* src = emb + v * E_;
  for (int e = threadIdx.x; e < E_; e += blockDim.x) {
    float x = src[e];
    wf[b * E_ + e] = x;
    wb[b * E_ + e] = (bf16_t)x;
  }
}

__global__ void k_init_h(const float* __restrict__ ih, float* __restrict__ hf,
                         bf16_t* __restrict__ hb) {
  int i = blockIdx.x * blockDim.x + threadIdx.x;
  if (i < B_ * H_) { float x = ih[i]; hf[i] = x; hb[i] = (bf16_t)x; }
}

// ------------------------------- GRU step ----------------------------------
// One workgroup, 512 threads = 16 wave32s. Wave w owns the 16x16 output tile
// (mbase=(w>>2)*16, jbase=(w&3)*16) through all three dependent stages so
// z_t / h_gru stay in registers across barriers. LDS carries h_gru and
// rt*word_emb (bf16, 64KB each) between stages.

__global__ void __launch_bounds__(512)
k_gru_step(const bf16_t* __restrict__ xseq,    // [T*B,E] bf16
           const bf16_t* __restrict__ wemb_b,  // [B,E] bf16
           const float*  __restrict__ wemb_f,  // [B,E] f32
           const bf16_t* __restrict__ Wih,     // [3H,E]
           const bf16_t* __restrict__ Whh,     // [3H,H]
           const float* __restrict__ b_ih, const float* __restrict__ b_hh,
           const bf16_t* __restrict__ ztW, const float* __restrict__ zt_b,
           const bf16_t* __restrict__ rtW, const float* __restrict__ rt_b,
           const bf16_t* __restrict__ htW, const float* __restrict__ ht_b,
           float*  __restrict__ h_f,           // [B,H] f32, in/out
           bf16_t* __restrict__ h_b,           // [B,H] bf16, in/out
           bf16_t* __restrict__ outs,          // [T*B,H] bf16 (rnn outputs)
           int t) {
  extern __shared__ char smem[];
  bf16_t* s_hgru = (bf16_t*)smem;                              // 64x512 bf16
  bf16_t* s_rtw  = (bf16_t*)(smem + (size_t)B_ * H_ * 2);      // 64x512 bf16

  const int tid   = threadIdx.x;
  const int lane  = tid & 31;
  const int wave  = tid >> 5;            // 0..15
  const int mbase = (wave >> 2) * 16;
  const int jbase = (wave & 3) * 16;
  const int j     = jbase + (lane & 15); // output column of this lane

  // ---- stage 1: gi = x@Wih^T, gh = h@Whh^T, GRU gates -> h_gru ----
  const bf16_t* xA = xseq + (t * B_ + mbase) * E_;
  const bf16_t* hA = h_b + mbase * H_;
  v8f ai0 = {}, ai1 = {}, ai2 = {}, ah0 = {}, ah1 = {}, ah2 = {};
#pragma unroll 4
  for (int kk = 0; kk < E_ / 32; ++kk) {
    const int k0 = kk * 32;
    v16bf ax = load_frag(xA, E_, lane, k0);
    v16bf ah = load_frag(hA, H_, lane, k0);
    v16bf b0 = load_frag(Wih + (0 * H_ + jbase) * E_, E_, lane, k0);
    v16bf b1 = load_frag(Wih + (1 * H_ + jbase) * E_, E_, lane, k0);
    v16bf b2 = load_frag(Wih + (2 * H_ + jbase) * E_, E_, lane, k0);
    ai0 = wmma_bf16(ax, b0, ai0);
    ai1 = wmma_bf16(ax, b1, ai1);
    ai2 = wmma_bf16(ax, b2, ai2);
    v16bf c0 = load_frag(Whh + (0 * H_ + jbase) * H_, H_, lane, k0);
    v16bf c1 = load_frag(Whh + (1 * H_ + jbase) * H_, H_, lane, k0);
    v16bf c2 = load_frag(Whh + (2 * H_ + jbase) * H_, H_, lane, k0);
    ah0 = wmma_bf16(ah, c0, ah0);
    ah1 = wmma_bf16(ah, c1, ah1);
    ah2 = wmma_bf16(ah, c2, ah2);
  }
  float hg[8];
#pragma unroll
  for (int v = 0; v < 8; ++v) {
    const int m = mbase + v + ((lane & 16) ? 8 : 0);
    float r = sigmoidf_((ai0[v] + b_ih[j]) + (ah0[v] + b_hh[j]));
    float z = sigmoidf_((ai1[v] + b_ih[H_ + j]) + (ah1[v] + b_hh[H_ + j]));
    float n = tanhf((ai2[v] + b_ih[2 * H_ + j]) +
                    r * (ah2[v] + b_hh[2 * H_ + j]));
    float hp = h_f[m * H_ + j];
    float g  = (1.0f - z) * n + z * hp;
    hg[v] = g;
    s_hgru[m * H_ + j] = (bf16_t)g;
    outs[(t * B_ + m) * H_ + j] = (bf16_t)g;   // rnn output for decode
  }
  __syncthreads();

  // ---- stage 2: zt/rt = sigmoid([wemb, h_gru] @ W^T), K = 1024 ----
  v8f az = {}, ar = {};
  const bf16_t* zB = ztW + jbase * (E_ + H_);
  const bf16_t* rB = rtW + jbase * (E_ + H_);
#pragma unroll 4
  for (int kk = 0; kk < (E_ + H_) / 32; ++kk) {
    const int k0 = kk * 32;
    v16bf a = (k0 < E_) ? load_frag(wemb_b + mbase * E_, E_, lane, k0)
                        : load_frag(s_hgru + mbase * H_, H_, lane, k0 - E_);
    v16bf bz = load_frag(zB, E_ + H_, lane, k0);
    v16bf br = load_frag(rB, E_ + H_, lane, k0);
    az = wmma_bf16(a, bz, az);
    ar = wmma_bf16(a, br, ar);
  }
  float zt[8];
#pragma unroll
  for (int v = 0; v < 8; ++v) {
    const int m = mbase + v + ((lane & 16) ? 8 : 0);
    zt[v] = sigmoidf_(az[v] + zt_b[j]);
    float rt = sigmoidf_(ar[v] + rt_b[j]);
    s_rtw[m * E_ + j] = (bf16_t)(rt * wemb_f[m * E_ + j]);
  }
  __syncthreads();

  // ---- stage 3: h_tilde = tanh([rt*wemb, h_gru] @ htW^T); h_new ----
  v8f at = {};
  const bf16_t* tB = htW + jbase * (E_ + H_);
#pragma unroll 4
  for (int kk = 0; kk < (E_ + H_) / 32; ++kk) {
    const int k0 = kk * 32;
    v16bf a = (k0 < E_) ? load_frag(s_rtw + mbase * E_, E_, lane, k0)
                        : load_frag(s_hgru + mbase * H_, H_, lane, k0 - E_);
    v16bf bt = load_frag(tB, E_ + H_, lane, k0);
    at = wmma_bf16(a, bt, at);
  }
#pragma unroll
  for (int v = 0; v < 8; ++v) {
    const int m = mbase + v + ((lane & 16) ? 8 : 0);
    float htl = tanhf(at[v] + ht_b[j]);
    float hn  = (1.0f - zt[v]) * hg[v] + zt[v] * htl;
    h_f[m * H_ + j] = hn;
    h_b[m * H_ + j] = (bf16_t)hn;
  }
}

// ------------------------------- decode GEMM -------------------------------
// out[4096,32000] = outs[4096,512] @ decW[32000,512]^T + dec_b.
// WG = 256 threads = 8 waves; tile = 128 rows x 128 cols. The 128x512 bf16
// A-tile (128 KB) is DMA'd into LDS by the Tensor Data Mover (one descriptor,
// issued by wave 0, synced with s_wait_tensorcnt); B rows stream from L2.
// Halving the grid in M (vs 64-row tiles) halves dec_W L2 traffic (~1 GB).

__global__ void __launch_bounds__(256)
k_decode(const bf16_t* __restrict__ outs, const bf16_t* __restrict__ decW,
         const float* __restrict__ dec_b, float* __restrict__ out) {
  extern __shared__ char smem[];
  bf16_t* As = (bf16_t*)smem;                 // 128 x 512 bf16 = 128 KB

  const int tid   = threadIdx.x;
  const int lane  = tid & 31;
  const int wave  = tid >> 5;                 // 0..7
  const int mtile = blockIdx.y;               // 0..31  (128 rows each)
  const int ctile = blockIdx.x;               // 0..249 (128 cols each)

#if defined(HAS_TDM)
  if (wave == 0) {
    // --- Tensor DMA descriptor (D#), 2D tile: 512 x 128 rows of bf16 ---
    const unsigned lds_base = (unsigned)(uintptr_t)(void*)As;
    const unsigned long long ga =
        (unsigned long long)(uintptr_t)(outs + (size_t)mtile * 128 * H_);
    u32x4 g0 = {};
    g0[0] = 1u;                                    // count=1 (valid user D#)
    g0[1] = lds_base;                              // lds_addr
    g0[2] = (unsigned)(ga & 0xffffffffu);          // global_addr[31:0]
    g0[3] = (unsigned)((ga >> 32) & 0x01ffffffu)   // global_addr[56:32]
            | 0x80000000u;                         // type=2 ("image")
    i32x8 g1 = {};
    g1[0] = (1 << 16);                             // data_size=1 -> 2 bytes
    g1[1] = (int)((H_ & 0xffff) << 16);            // tensor_dim0 = 512
    g1[2] = (int)(((unsigned)(T_ * B_) & 0xffff) << 16); // tensor_dim1 = 4096
    g1[3] = (int)((H_ & 0xffff) << 16);            // tile_dim0 = 512
    g1[4] = 128;                                   // tile_dim1 = 128 rows
    g1[5] = H_;                                    // tensor_dim0_stride = 512
    i32x4 g2 = {}, g3 = {};
#if __clang_major__ >= 23
    __builtin_amdgcn_tensor_load_to_lds(g0, g1, g2, g3,
                                        (i32x8){0, 0, 0, 0, 0, 0, 0, 0}, 0);
#else
    __builtin_amdgcn_tensor_load_to_lds(g0, g1, g2, g3, 0);
#endif
    __builtin_amdgcn_s_wait_tensorcnt(0);
  }
#else
  {  // fallback: cooperative 128 KB copy, b128 both sides
    const v8bf* src = (const v8bf*)(outs + (size_t)mtile * 128 * H_);
    v8bf* dst = (v8bf*)As;
#pragma unroll
    for (int i = 0; i < 32; ++i) dst[i * 256 + tid] = src[i * 256 + tid];
  }
#endif
  __syncthreads();

  // wave (mi, ni): mi = wave>>1 -> 32 rows, ni = wave&1 -> 64 cols
  const int r0      = (wave >> 1) * 32;
  const int colbase = ctile * 128 + (wave & 1) * 64;

  v8f acc[2][4] = {};
#pragma unroll 4
  for (int kk = 0; kk < H_ / 32; ++kk) {
    const int k0 = kk * 32;
    v16bf a0 = load_frag(As + (r0 +  0) * H_, H_, lane, k0);
    v16bf a1 = load_frag(As + (r0 + 16) * H_, H_, lane, k0);
#pragma unroll
    for (int nb = 0; nb < 4; ++nb) {
      v16bf b = load_frag(decW + (long long)(colbase + nb * 16) * H_, H_,
                          lane, k0);
      acc[0][nb] = wmma_bf16(a0, b, acc[0][nb]);
      acc[1][nb] = wmma_bf16(a1, b, acc[1][nb]);
    }
  }
  const int n = lane & 15;
#pragma unroll
  for (int mb = 0; mb < 2; ++mb) {
#pragma unroll
    for (int nb = 0; nb < 4; ++nb) {
      const int col = colbase + nb * 16 + n;
      const float bias = dec_b[col];
#pragma unroll
      for (int v = 0; v < 8; ++v) {
        const long long m = (long long)mtile * 128 + r0 + mb * 16 + v +
                            ((lane & 16) ? 8 : 0);
        out[m * V_ + col] = acc[mb][nb][v] + bias;
      }
    }
  }
}

__global__ void k_copy_hidden(const float* __restrict__ hf,
                              float* __restrict__ out) {
  int i = blockIdx.x * blockDim.x + threadIdx.x;
  if (i < B_ * H_) out[i] = hf[i];
}

// ------------------------------- launcher ----------------------------------

extern "C" void kernel_launch(void* const* d_in, const int* in_sizes, int n_in,
                              void* d_out, int out_size, void* d_ws,
                              size_t ws_size, hipStream_t stream) {
  const float* emb   = (const float*)d_in[0];
  const float* W_ih  = (const float*)d_in[1];
  const float* W_hh  = (const float*)d_in[2];
  const float* b_ih  = (const float*)d_in[3];
  const float* b_hh  = (const float*)d_in[4];
  const float* zt_W  = (const float*)d_in[5];
  const float* zt_b  = (const float*)d_in[6];
  const float* rt_W  = (const float*)d_in[7];
  const float* rt_b  = (const float*)d_in[8];
  const float* ht_W  = (const float*)d_in[9];
  const float* ht_b  = (const float*)d_in[10];
  const float* dec_W = (const float*)d_in[11];
  const float* dec_b = (const float*)d_in[12];
  const float* init_hidden = (const float*)d_in[13];
  const int*   word  = (const int*)d_in[14];
  const int*   seq   = (const int*)d_in[15];

  char* ws = (char*)d_ws;
  size_t o = 0;
  auto alloc = [&](size_t bytes) -> char* {
    char* p = ws + o;
    o += (bytes + 255) & ~(size_t)255;
    return p;
  };
  bf16_t* seq_b  = (bf16_t*)alloc((size_t)T_ * B_ * E_ * 2);     // 4 MB
  bf16_t* wemb_b = (bf16_t*)alloc((size_t)B_ * E_ * 2);
  float*  wemb_f = (float*) alloc((size_t)B_ * E_ * 4);
  bf16_t* Wih_b  = (bf16_t*)alloc((size_t)3 * H_ * E_ * 2);
  bf16_t* Whh_b  = (bf16_t*)alloc((size_t)3 * H_ * H_ * 2);
  bf16_t* ztW_b  = (bf16_t*)alloc((size_t)H_ * (E_ + H_) * 2);
  bf16_t* rtW_b  = (bf16_t*)alloc((size_t)E_ * (E_ + H_) * 2);
  bf16_t* htW_b  = (bf16_t*)alloc((size_t)H_ * (E_ + H_) * 2);
  bf16_t* decW_b = (bf16_t*)alloc((size_t)V_ * H_ * 2);          // 32 MB
  bf16_t* outs_b = (bf16_t*)alloc((size_t)T_ * B_ * H_ * 2);     // 4 MB
  float*  h_f    = (float*) alloc((size_t)B_ * H_ * 4);
  bf16_t* h_b    = (bf16_t*)alloc((size_t)B_ * H_ * 2);

  // one-time (per call) bf16 conversions + gathers
  k_f32_to_bf16<<<1024, 256, 0, stream>>>(W_ih, Wih_b, 3 * H_ * E_);
  k_f32_to_bf16<<<1024, 256, 0, stream>>>(W_hh, Whh_b, 3 * H_ * H_);
  k_f32_to_bf16<<<1024, 256, 0, stream>>>(zt_W, ztW_b, H_ * (E_ + H_));
  k_f32_to_bf16<<<1024, 256, 0, stream>>>(rt_W, rtW_b, E_ * (E_ + H_));
  k_f32_to_bf16<<<1024, 256, 0, stream>>>(ht_W, htW_b, H_ * (E_ + H_));
  k_f32_to_bf16<<<4096, 256, 0, stream>>>(dec_W, decW_b, V_ * H_);
  k_gather_seq<<<T_ * B_, 256, 0, stream>>>(emb, seq, seq_b);
  k_gather_word<<<B_, 256, 0, stream>>>(emb, word, wemb_b, wemb_f);
  k_init_h<<<(B_ * H_ + 255) / 256, 256, 0, stream>>>(init_hidden, h_f, h_b);

  // sequential recurrence: one WG (16 waves) per step, weights hot in L2
  const int gru_lds = 2 * B_ * H_ * (int)sizeof(bf16_t);  // 128 KB
  for (int t = 0; t < T_; ++t) {
    k_gru_step<<<1, 512, gru_lds, stream>>>(
        seq_b, wemb_b, wemb_f, Wih_b, Whh_b, b_ih, b_hh,
        ztW_b, zt_b, rtW_b, rt_b, htW_b, ht_b, h_f, h_b, outs_b, t);
  }

  // decode: 134 GFLOP bf16 GEMM, 128x128 tiles, A tile via TDM into LDS
  dim3 dgrid(V_ / 128, (T_ * B_) / 128);                  // 250 x 32
  const int dec_lds = 128 * H_ * (int)sizeof(bf16_t);     // 128 KB
  k_decode<<<dgrid, 256, dec_lds, stream>>>(outs_b, decW_b, dec_b,
                                            (float*)d_out);
  // hidden state output appended after decoded logits
  k_copy_hidden<<<(B_ * H_ + 255) / 256, 256, 0, stream>>>(
      h_f, (float*)d_out + (size_t)T_ * B_ * V_);
}
